// Colorizer_53309134078077
// MI455X (gfx1250) — matrix operand order repaired and checked by reading, not compiled, not run on hardware
//
#include <hip/hip_runtime.h>
#include <hip/hip_bf16.h>

#define H_ 270
#define W_ 512
#define B_ 4
#define HW_ (H_*W_)
#define NPIX (B_*HW_)
#define PH_ (H_+1)          // pooled height 271
#define PW_ (W_+1)          // pooled width  513
#define NPOOL (B_*8*PH_*PW_)
#define NPART 512
#define PI_F 3.14159265358979f

typedef _Float16 v16h __attribute__((ext_vector_type(16)));
typedef _Float16 v8h  __attribute__((ext_vector_type(8)));
typedef float    v8f  __attribute__((ext_vector_type(8)));
typedef unsigned int u32x4 __attribute__((ext_vector_type(4)));
typedef int      i32x4 __attribute__((ext_vector_type(4)));
typedef int      i32x8 __attribute__((ext_vector_type(8)));

// Tensor Data Mover availability (probe-confirmed builtins; arity differs per toolchain)
#if defined(__has_builtin)
# if __has_builtin(__builtin_amdgcn_tensor_load_to_lds) && __has_builtin(__builtin_amdgcn_s_wait_tensorcnt)
#  define USE_TDM 1
# endif
#endif
#ifndef USE_TDM
# define USE_TDM 0
#endif

__device__ __forceinline__ int clampi(int v, int lo, int hi) {
    return v < lo ? lo : (v > hi ? hi : v);
}
__device__ __forceinline__ int reflecti(int v, int n) {
    return v < 0 ? -v : (v >= n ? 2*n - 2 - v : v);
}

// ---------------------------------------------------------------------------
// 1) dense SIFT: per-pixel gradient orientation histogram (8 bins)
// ---------------------------------------------------------------------------
__global__ __launch_bounds__(256) void sift_ang_kernel(const float* __restrict__ x,
                                                       float* __restrict__ ang) {
    int idx = blockIdx.x * 256 + threadIdx.x;
    if (idx >= NPIX) return;
    int j = idx % W_; int t = idx / W_; int i = t % H_; int b = t / H_;
    const float* xp = x + (size_t)b * HW_;
    float xl = xp[i*W_ + clampi(j-1, 0, W_-1)];
    float xr = xp[i*W_ + clampi(j+1, 0, W_-1)];
    float xu = xp[clampi(i-1, 0, H_-1)*W_ + j];
    float xd = xp[clampi(i+1, 0, H_-1)*W_ + j];
    float gx = xr - xl;
    float gy = xd - xu;
    float mag = sqrtf(gx*gx + gy*gy + 1e-10f);
    float ori = atan2f(gy, gx + 1e-10f) + 2.0f * PI_F;
    float ob  = 8.0f * ori / (2.0f * PI_F);           // in (4, 12]
    float b0f = floorf(ob);
    float wo1 = ob - b0f;
    int ib0 = ((int)b0f) & 7;
    int ib1 = (ib0 + 1) & 7;
    float w0 = (1.0f - wo1) * mag;
    float w1 = wo1 * mag;
    size_t base = ((size_t)b * 8) * HW_ + (size_t)i * W_ + j;
    #pragma unroll
    for (int o = 0; o < 8; ++o) {
        float v = (o == ib0) ? w0 : ((o == ib1) ? w1 : 0.0f);
        ang[base + (size_t)o * HW_] = v;
    }
}

// ---------------------------------------------------------------------------
// 2) 4x4 box pooling, pad=2 -> 271x513 per (b,bin)
// ---------------------------------------------------------------------------
__global__ __launch_bounds__(256) void sift_pool_kernel(const float* __restrict__ ang,
                                                        float* __restrict__ pooled) {
    int idx = blockIdx.x * 256 + threadIdx.x;
    if (idx >= NPOOL) return;
    int j = idx % PW_; int t = idx / PW_; int i = t % PH_; int bo = t / PH_;
    const float* ap = ang + (size_t)bo * HW_;
    float s = 0.0f;
    #pragma unroll
    for (int u = 0; u < 4; ++u) {
        int y = i - 2 + u;
        if (y < 0 || y >= H_) continue;
        #pragma unroll
        for (int v = 0; v < 4; ++v) {
            int xx = j - 2 + v;
            if (xx < 0 || xx >= W_) continue;
            s += ap[y*W_ + xx];
        }
    }
    pooled[idx] = s * (1.0f / 16.0f);
}

// ---------------------------------------------------------------------------
// 3) gather 16 shifts x 8 bins, normalize chain, emit NHWC fp16 feature map
// ---------------------------------------------------------------------------
__global__ __launch_bounds__(256) void sift_final_kernel(const float* __restrict__ pooled,
                                                         _Float16* __restrict__ feat) {
    int idx = blockIdx.x * 256 + threadIdx.x;
    if (idx >= NPIX) return;
    int j = idx % W_; int t = idx / W_; int i = t % H_; int b = t / H_;
    const float* pb = pooled + (size_t)b * 8 * PH_ * PW_;
    auto P = [&](int o, int di, int dj) -> float {
        int y = i + di - 1, xx = j + dj - 1;
        if (y < 0 || y > H_ || xx < 0 || xx > W_) return 0.0f;   // pooled is 271x513
        return pb[((size_t)o * PH_ + y) * PW_ + xx];
    };
    float s2 = 0.0f;
    for (int o = 0; o < 8; ++o)
        for (int di = 0; di < 4; ++di)
            for (int dj = 0; dj < 4; ++dj) { float p = P(o,di,dj); s2 += p*p; }
    float inv2 = 1.0f / fmaxf(sqrtf(s2), 1e-12f);
    float s2b = 0.0f, sc = 0.0f;
    for (int o = 0; o < 8; ++o)
        for (int di = 0; di < 4; ++di)
            for (int dj = 0; dj < 4; ++dj) {
                float c = fminf(P(o,di,dj) * inv2, 0.2f);
                s2b += c*c; sc += c;
            }
    float inv2b = 1.0f / fmaxf(sqrtf(s2b), 1e-12f);
    float l1    = sc * inv2b;
    float invl1 = 1.0f / fmaxf(l1, 1e-12f);
    _Float16* fp = feat + (size_t)idx * 128;
    for (int o = 0; o < 8; ++o)
        for (int di = 0; di < 4; ++di)
            for (int dj = 0; dj < 4; ++dj) {
                float c = fminf(P(o,di,dj) * inv2, 0.2f);
                float w = c * inv2b;
                fp[o*16 + di*4 + dj] = (_Float16)sqrtf(w * invl1 + 1e-10f);
            }
}

// ---------------------------------------------------------------------------
// 4) weight packing into WMMA B-fragment lane layout
//    B 32x16 fp16 frag: lane<16 -> col N=lane, K=0..7,16..23; lane>=16 -> col N=lane-16, K=8..15,24..31
// ---------------------------------------------------------------------------
__global__ __launch_bounds__(256) void pack_w1_kernel(const float* __restrict__ w,
                                                      _Float16* __restrict__ wp) {
    int idx = blockIdx.x * 256 + threadIdx.x;
    if (idx >= 25*4*2*32*16) return;
    int q    = idx & 15;
    int lane = (idx >> 4) & 31;
    int nt   = (idx >> 9) & 1;
    int kc   = (idx >> 10) & 3;
    int tap  = idx >> 12;                    // 0..24
    int ky = tap / 5, kx = tap % 5;
    int n    = lane & 15;
    int koff = ((lane >> 4) << 3) + (q & 7) + ((q >> 3) << 4);       // 0..31
    int c = kc * 32 + koff;
    int o = nt * 16 + n;
    wp[idx] = (_Float16)w[((o * 128 + c) * 5 + ky) * 5 + kx];
}

__global__ __launch_bounds__(256) void pack_w2_kernel(const float* __restrict__ w,
                                                      _Float16* __restrict__ wp) {
    int idx = blockIdx.x * 256 + threadIdx.x;
    if (idx >= 9*32*16) return;
    int q    = idx & 15;
    int lane = (idx >> 4) & 31;
    int tap  = idx >> 9;                     // 0..8
    int ky = tap / 3, kx = tap % 3;
    int n    = lane & 15;
    int koff = ((lane >> 4) << 3) + (q & 7) + ((q >> 3) << 4);       // c in 0..31
    float v = (n < 8) ? w[((n * 32 + koff) * 3 + ky) * 3 + kx] : 0.0f;
    wp[idx] = (_Float16)v;
}

// ---------------------------------------------------------------------------
// 5) conv1 128->32, 5x5, pad 2, ReLU — implicit GEMM on v_wmma_f32_16x16x32_f16
//    one wave32 = 32 pixels x 32 out channels (two M tiles share each B frag)
// ---------------------------------------------------------------------------
__global__ __launch_bounds__(128) void conv1_wmma_kernel(const _Float16* __restrict__ feat,
                                                         const _Float16* __restrict__ wpack,
                                                         const float* __restrict__ bias,
                                                         _Float16* __restrict__ act1) {
    const int wave = (blockIdx.x << 2) | (threadIdx.x >> 5);
    const int lane = threadIdx.x & 31;
    const int m    = lane & 15;
    const int half = lane >> 4;
    int b   = wave / (H_ * 16);
    int rem = wave - b * (H_ * 16);
    int i   = rem >> 4;
    int j0  = (rem & 15) << 5;               // 32-pixel supertile

    v8f accA0 = {}, accA1 = {}, accB0 = {}, accB1 = {};
    float bias0 = bias[m];
    float bias1 = bias[16 + m];

    for (int ky = 0; ky < 5; ++ky) {
        int iy = i + ky - 2;
        bool rowOK = (iy >= 0) && (iy < H_);
        const _Float16* rowbase = feat + ((size_t)(b * H_ + iy) * W_) * 128;
        if (rowOK) __builtin_prefetch(rowbase + (size_t)(j0 + m) * 128, 0, 0);
        for (int kx = 0; kx < 5; ++kx) {
            int jxA = j0 + m + kx - 2;
            int jxB = jxA + 16;
            bool okA = rowOK && (jxA >= 0) && (jxA < W_);
            bool okB = rowOK && (jxB < W_);
            const _Float16* apA = rowbase + (size_t)jxA * 128 + half * 8;
            const _Float16* apB = apA + (size_t)16 * 128;
            const int tap = ky * 5 + kx;
            #pragma unroll
            for (int kc = 0; kc < 4; ++kc) {
                v16h a0 = {}, a1 = {};
                if (okA) {
                    v8h lo = *(const v8h*)(apA + kc * 32);
                    v8h hi = *(const v8h*)(apA + kc * 32 + 16);
                    a0 = __builtin_shufflevector(lo, hi, 0,1,2,3,4,5,6,7,8,9,10,11,12,13,14,15);
                }
                if (okB) {
                    v8h lo = *(const v8h*)(apB + kc * 32);
                    v8h hi = *(const v8h*)(apB + kc * 32 + 16);
                    a1 = __builtin_shufflevector(lo, hi, 0,1,2,3,4,5,6,7,8,9,10,11,12,13,14,15);
                }
                const v16h bf0 = *(const v16h*)(wpack + ((size_t)((tap*4 + kc)*2 + 0))*512 + lane*16);
                const v16h bf1 = *(const v16h*)(wpack + ((size_t)((tap*4 + kc)*2 + 1))*512 + lane*16);
                accA0 = __builtin_amdgcn_wmma_f32_16x16x32_f16(false, a0, false, bf0, (short)0, accA0, false, false);
                accA1 = __builtin_amdgcn_wmma_f32_16x16x32_f16(false, a0, false, bf1, (short)0, accA1, false, false);
                accB0 = __builtin_amdgcn_wmma_f32_16x16x32_f16(false, a1, false, bf0, (short)0, accB0, false, false);
                accB1 = __builtin_amdgcn_wmma_f32_16x16x32_f16(false, a1, false, bf1, (short)0, accB1, false, false);
            }
        }
    }
    // C/D layout: lane L, vgpr r -> pixel (tile + r + 8*half), channel ntile*16 + (L&15)
    _Float16* obase = act1 + ((size_t)(b * H_ + i) * W_ + j0) * 32;
    #pragma unroll
    for (int r = 0; r < 8; ++r) {
        int pix = r + half * 8;
        float v0 = accA0[r] + bias0; v0 = v0 > 0.0f ? v0 : 0.0f;
        float v1 = accA1[r] + bias1; v1 = v1 > 0.0f ? v1 : 0.0f;
        float v2 = accB0[r] + bias0; v2 = v2 > 0.0f ? v2 : 0.0f;
        float v3 = accB1[r] + bias1; v3 = v3 > 0.0f ? v3 : 0.0f;
        obase[(size_t)pix * 32 + m]             = (_Float16)v0;
        obase[(size_t)pix * 32 + 16 + m]        = (_Float16)v1;
        obase[(size_t)(pix + 16) * 32 + m]      = (_Float16)v2;
        obase[(size_t)(pix + 16) * 32 + 16 + m] = (_Float16)v3;
    }
}

// ---------------------------------------------------------------------------
// 6) conv2 32->8, 3x3, pad 1, ReLU — WMMA with B fragments staged in LDS.
//    Weights (9216 B) are DMA'd by the Tensor Data Mover when available.
// ---------------------------------------------------------------------------
__global__ __launch_bounds__(128) void conv2_wmma_kernel(const _Float16* __restrict__ act1,
                                                         const _Float16* __restrict__ wpack2,
                                                         const float* __restrict__ bias,
                                                         _Float16* __restrict__ act2) {
    __shared__ _Float16 lds_w2[9 * 512];     // 9216 bytes

#if USE_TDM
    if (threadIdx.x == 0) {
        // Tensor DMA descriptor (CDNA5 ISA):
        // 1D tile: 1152 elements x 8 bytes = 9216 B, global -> LDS.
        unsigned long long ga = (unsigned long long)(size_t)wpack2;
        unsigned int lds = (unsigned int)(size_t)&lds_w2[0];   // low 32 bits = LDS offset
        u32x4 g0;
        g0[0] = 1u;                                            // count=1 (valid user D#)
        g0[1] = lds;                                           // lds_addr
        g0[2] = (unsigned int)ga;                              // global_addr[31:0]
        g0[3] = (unsigned int)((ga >> 32) & 0x01FFFFFFu)       // global_addr[56:32]
              | (2u << 30);                                    // type=2 ("image")
        i32x8 g1;
        g1[0] = (int)(3u << 16);                               // data_size=3 (8B), mask=0
        g1[1] = (int)(1152u << 16);                            // tensor_dim0[15:0] -> [31:16]
        g1[2] = (int)(1u << 16);                               // tensor_dim1=1
        g1[3] = (int)(1152u << 16);                            // tile_dim0=1152
        g1[4] = 1;                                             // tile_dim1=1, tile_dim2=0
        g1[5] = 1152;                                          // tensor_dim0_stride lo32
        g1[6] = 0;
        g1[7] = 0;
        i32x4 gz = {0, 0, 0, 0};
#if __has_include(<hip/amd_detail/amd_gfx1250_TDM.h>)
        i32x8 gz8 = {0, 0, 0, 0, 0, 0, 0, 0};
        __builtin_amdgcn_tensor_load_to_lds(g0, g1, gz, gz, gz8, 0);
#else
        __builtin_amdgcn_tensor_load_to_lds(g0, g1, gz, gz, 0);
#endif
        __builtin_amdgcn_s_wait_tensorcnt(0);
    }
#else
    for (int t = threadIdx.x; t < 9 * 512 / 8; t += 128)
        ((v8h*)lds_w2)[t] = ((const v8h*)wpack2)[t];
#endif
    __syncthreads();

    const int wave = (blockIdx.x << 2) | (threadIdx.x >> 5);
    const int lane = threadIdx.x & 31;
    const int m    = lane & 15;
    const int half = lane >> 4;
    int b   = wave / (H_ * 32);
    int rem = wave - b * (H_ * 32);
    int i   = rem >> 5;
    int j0  = (rem & 31) << 4;

    v8f acc = {};
    float bz = (m < 8) ? bias[m] : 0.0f;

    for (int ky = 0; ky < 3; ++ky) {
        int iy = i + ky - 1;
        bool rowOK = (iy >= 0) && (iy < H_);
        for (int kx = 0; kx < 3; ++kx) {
            int jx = j0 + m + kx - 1;
            bool ok = rowOK && (jx >= 0) && (jx < W_);
            v16h a = {};
            if (ok) {
                const _Float16* ap = act1 + ((size_t)(b * H_ + iy) * W_ + jx) * 32 + half * 8;
                v8h lo = *(const v8h*)ap;
                v8h hi = *(const v8h*)(ap + 16);
                a = __builtin_shufflevector(lo, hi, 0,1,2,3,4,5,6,7,8,9,10,11,12,13,14,15);
            }
            const v16h bf = *(const v16h*)&lds_w2[(ky*3 + kx)*512 + lane*16];   // ds_load_b128
            acc = __builtin_amdgcn_wmma_f32_16x16x32_f16(false, a, false, bf, (short)0, acc, false, false);
        }
    }
    if (m < 8) {
        _Float16* obase = act2 + ((size_t)(b * H_ + i) * W_ + j0) * 8;
        #pragma unroll
        for (int r = 0; r < 8; ++r) {
            int pix = r + half * 8;
            float v = acc[r] + bz; v = v > 0.0f ? v : 0.0f;
            obase[(size_t)pix * 8 + m] = (_Float16)v;
        }
    }
}

// ---------------------------------------------------------------------------
// 7) conv3 8->2, 5x5, pad 2 (tiny; plain VALU)
// ---------------------------------------------------------------------------
__global__ __launch_bounds__(256) void conv3_kernel(const _Float16* __restrict__ act2,
                                                    const float* __restrict__ w,
                                                    const float* __restrict__ bias,
                                                    float* __restrict__ f3) {
    int idx = blockIdx.x * 256 + threadIdx.x;
    if (idx >= NPIX) return;
    int j = idx % W_; int t = idx / W_; int i = t % H_; int b = t / H_;
    float a0 = bias[0], a1 = bias[1];
    for (int ky = 0; ky < 5; ++ky) {
        int iy = i + ky - 2;
        if (iy < 0 || iy >= H_) continue;
        for (int kx = 0; kx < 5; ++kx) {
            int jx = j + kx - 2;
            if (jx < 0 || jx >= W_) continue;
            const _Float16* ap = act2 + ((size_t)(b * H_ + iy) * W_ + jx) * 8;
            #pragma unroll
            for (int c = 0; c < 8; ++c) {
                float av = (float)ap[c];
                a0 += av * w[((0*8 + c)*5 + ky)*5 + kx];
                a1 += av * w[((1*8 + c)*5 + ky)*5 + kx];
            }
        }
    }
    f3[((size_t)b*2 + 0)*HW_ + (size_t)i*W_ + j] = a0;
    f3[((size_t)b*2 + 1)*HW_ + (size_t)i*W_ + j] = a1;
}

// ---------------------------------------------------------------------------
// 8) joint bilateral blur 5x5 (guidance = concat(x,x) -> cd = 4*dx^2)
// ---------------------------------------------------------------------------
__global__ __launch_bounds__(256) void bilateral_kernel(const float* __restrict__ f3,
                                                        const float* __restrict__ x,
                                                        float* __restrict__ blur) {
    int idx = blockIdx.x * 256 + threadIdx.x;
    if (idx >= NPIX) return;
    int j = idx % W_; int t = idx / W_; int i = t % H_; int b = t / H_;
    const float* xb = x + (size_t)b * HW_;
    float g = xb[i*W_ + j];
    float g1[5]; float ssum = 0.0f;
    #pragma unroll
    for (int q = 0; q < 5; ++q) {
        float d = (float)(q - 2);
        g1[q] = expf(-(d*d) / (2.0f * 1.5f * 1.5f));
        ssum += g1[q];
    }
    #pragma unroll
    for (int q = 0; q < 5; ++q) g1[q] /= ssum;
    const float* f0 = f3 + (size_t)b * 2 * HW_;
    const float* f1 = f0 + HW_;
    float n0 = 0.0f, n1 = 0.0f, den = 0.0f;
    for (int dy = 0; dy < 5; ++dy) {
        int yy = reflecti(i + dy - 2, H_);
        for (int dx = 0; dx < 5; ++dx) {
            int xx = reflecti(j + dx - 2, W_);
            float gw = xb[yy*W_ + xx];
            float d = gw - g;
            float k = g1[dy] * g1[dx] * expf(-50.0f * (4.0f * d * d));
            n0 += f0[yy*W_ + xx] * k;
            n1 += f1[yy*W_ + xx] * k;
            den += k;
        }
    }
    blur[((size_t)b*2 + 0)*HW_ + (size_t)i*W_ + j] = n0 / den;
    blur[((size_t)b*2 + 1)*HW_ + (size_t)i*W_ + j] = n1 / den;
}

// ---------------------------------------------------------------------------
// 9) global min/max of u and v planes (two-stage, deterministic)
// ---------------------------------------------------------------------------
__global__ __launch_bounds__(256) void reduce_minmax1(const float* __restrict__ blur,
                                                      float* __restrict__ partial) {
    __shared__ float s0[256], s1[256], s2[256], s3[256];
    int tid = threadIdx.x;
    float mn0 = 3.4e38f, mx0 = -3.4e38f, mn1 = 3.4e38f, mx1 = -3.4e38f;
    for (int k = blockIdx.x * 256 + tid; k < NPIX; k += NPART * 256) {
        int b = k / HW_, r = k - b * HW_;
        float u = blur[((size_t)b*2 + 0)*HW_ + r];
        float v = blur[((size_t)b*2 + 1)*HW_ + r];
        mn0 = fminf(mn0, u); mx0 = fmaxf(mx0, u);
        mn1 = fminf(mn1, v); mx1 = fmaxf(mx1, v);
    }
    s0[tid] = mn0; s1[tid] = mx0; s2[tid] = mn1; s3[tid] = mx1;
    __syncthreads();
    for (int off = 128; off > 0; off >>= 1) {
        if (tid < off) {
            s0[tid] = fminf(s0[tid], s0[tid+off]);
            s1[tid] = fmaxf(s1[tid], s1[tid+off]);
            s2[tid] = fminf(s2[tid], s2[tid+off]);
            s3[tid] = fmaxf(s3[tid], s3[tid+off]);
        }
        __syncthreads();
    }
    if (tid == 0) {
        partial[blockIdx.x*4 + 0] = s0[0];
        partial[blockIdx.x*4 + 1] = s1[0];
        partial[blockIdx.x*4 + 2] = s2[0];
        partial[blockIdx.x*4 + 3] = s3[0];
    }
}

__global__ __launch_bounds__(256) void reduce_minmax2(const float* __restrict__ partial,
                                                      float* __restrict__ mm) {
    __shared__ float s0[256], s1[256], s2[256], s3[256];
    int tid = threadIdx.x;
    float mn0 = 3.4e38f, mx0 = -3.4e38f, mn1 = 3.4e38f, mx1 = -3.4e38f;
    for (int k = tid; k < NPART; k += 256) {
        mn0 = fminf(mn0, partial[k*4 + 0]);
        mx0 = fmaxf(mx0, partial[k*4 + 1]);
        mn1 = fminf(mn1, partial[k*4 + 2]);
        mx1 = fmaxf(mx1, partial[k*4 + 3]);
    }
    s0[tid] = mn0; s1[tid] = mx0; s2[tid] = mn1; s3[tid] = mx1;
    __syncthreads();
    for (int off = 128; off > 0; off >>= 1) {
        if (tid < off) {
            s0[tid] = fminf(s0[tid], s0[tid+off]);
            s1[tid] = fmaxf(s1[tid], s1[tid+off]);
            s2[tid] = fminf(s2[tid], s2[tid+off]);
            s3[tid] = fmaxf(s3[tid], s3[tid+off]);
        }
        __syncthreads();
    }
    if (tid == 0) { mm[0]=s0[0]; mm[1]=s1[0]; mm[2]=s2[0]; mm[3]=s3[0]; }
}

// ---------------------------------------------------------------------------
// 10) normalize U/V + YUV -> RGB
// ---------------------------------------------------------------------------
__global__ __launch_bounds__(256) void finalize_kernel(const float* __restrict__ blur,
                                                       const float* __restrict__ x,
                                                       const float* __restrict__ mm,
                                                       float* __restrict__ out) {
    int idx = blockIdx.x * 256 + threadIdx.x;
    if (idx >= NPIX) return;
    int b = idx / HW_, r = idx - b * HW_;
    float umin = mm[0], umax = mm[1], vmin = mm[2], vmax = mm[3];
    float u = blur[((size_t)b*2 + 0)*HW_ + r];
    float v = blur[((size_t)b*2 + 1)*HW_ + r];
    u = fminf(fmaxf((u - umin) / (umax - umin + 1e-6f), 0.0f), 1.0f) * (0.436f*2.0f) - 0.436f;
    v = fminf(fmaxf((v - vmin) / (vmax - vmin + 1e-6f), 0.0f), 1.0f) * (0.615f*2.0f) - 0.615f;
    float y = x[(size_t)b*HW_ + r];
    out[((size_t)b*3 + 0)*HW_ + r] = y + 1.14f * v;
    out[((size_t)b*3 + 1)*HW_ + r] = y - 0.396f * u - 0.581f * v;
    out[((size_t)b*3 + 2)*HW_ + r] = y + 2.029f * u;
}

// ---------------------------------------------------------------------------
// launch
// ---------------------------------------------------------------------------
static inline size_t alignup(size_t x) { return (x + 255) & ~(size_t)255; }

extern "C" void kernel_launch(void* const* d_in, const int* in_sizes, int n_in,
                              void* d_out, int out_size, void* d_ws, size_t ws_size,
                              hipStream_t stream) {
    const float* x       = (const float*)d_in[0];
    const float* conv1_w = (const float*)d_in[1];
    const float* conv1_b = (const float*)d_in[2];
    const float* conv2_w = (const float*)d_in[3];
    const float* conv2_b = (const float*)d_in[4];
    const float* conv3_w = (const float*)d_in[5];
    const float* conv3_b = (const float*)d_in[6];
    float* out = (float*)d_out;

    // workspace carve-out
    char* w = (char*)d_ws;
    size_t off = 0;
    float*    ang    = (float*)(w + off);    off = alignup(off + (size_t)8*NPIX*4);
    float*    pooled = (float*)(w + off);    off = alignup(off + (size_t)NPOOL*4);
    _Float16* feat   = (_Float16*)(w + off); off = alignup(off + (size_t)NPIX*128*2);
    _Float16* wpack1 = (_Float16*)(w + off); off = alignup(off + (size_t)25*4*2*32*16*2);
    _Float16* act1   = (_Float16*)(w + off); off = alignup(off + (size_t)NPIX*32*2);
    _Float16* wpack2 = (_Float16*)(w + off); off = alignup(off + (size_t)9*32*16*2);
    _Float16* act2   = (_Float16*)(w + off); off = alignup(off + (size_t)NPIX*8*2);
    float*    f3     = (float*)(w + off);    off = alignup(off + (size_t)2*NPIX*4);
    float*    blur   = (float*)(w + off);    off = alignup(off + (size_t)2*NPIX*4);
    float*    part   = (float*)(w + off);    off = alignup(off + (size_t)NPART*4*4);
    float*    mm     = (float*)(w + off);    off = alignup(off + 4*4);

    const int gPix   = (NPIX  + 255) / 256;
    const int gPool  = (NPOOL + 255) / 256;
    const int gConv1 = (B_ * H_ * 16) / 4;    // 32-pixel supertiles, 4 waves/block
    const int gConv2 = (B_ * H_ * 32) / 4;    // 16-pixel tiles, 4 waves/block

    // weight packing (tiny)
    pack_w1_kernel<<<(25*4*2*32*16 + 255)/256, 256, 0, stream>>>(conv1_w, wpack1);
    pack_w2_kernel<<<(9*32*16 + 255)/256, 256, 0, stream>>>(conv2_w, wpack2);

    // dense SIFT
    sift_ang_kernel  <<<gPix, 256, 0, stream>>>(x, ang);
    sift_pool_kernel <<<gPool, 256, 0, stream>>>(ang, pooled);
    sift_final_kernel<<<gPix, 256, 0, stream>>>(pooled, feat);

    // conv stack (WMMA implicit GEMM for conv1/conv2)
    conv1_wmma_kernel<<<gConv1, 128, 0, stream>>>(feat, wpack1, conv1_b, act1);
    conv2_wmma_kernel<<<gConv2, 128, 0, stream>>>(act1, wpack2, conv2_b, act2);
    conv3_kernel     <<<gPix, 256, 0, stream>>>(act2, conv3_w, conv3_b, f3);

    // bilateral + normalization + color transform
    bilateral_kernel<<<gPix, 256, 0, stream>>>(f3, x, blur);
    reduce_minmax1<<<NPART, 256, 0, stream>>>(blur, part);
    reduce_minmax2<<<1, 256, 0, stream>>>(part, mm);
    finalize_kernel<<<gPix, 256, 0, stream>>>(blur, x, mm, out);
}
